// AttentionMechanism_30975304139500
// MI455X (gfx1250) — matrix operand order repaired and checked by reading, not compiled, not run on hardware
//
#include <hip/hip_runtime.h>
#include <stdint.h>

#define B_ 64
#define S_ 1024
#define ED 1024
#define DD 512
#define NEG_INF -1000000000.0f

typedef __attribute__((ext_vector_type(16))) __bf16 v16bf;
typedef __attribute__((ext_vector_type(8)))  __bf16 v8bf;
typedef __attribute__((ext_vector_type(4)))  __bf16 v4bf;
typedef __attribute__((ext_vector_type(8)))  float  v8f;
typedef __attribute__((ext_vector_type(4)))  float  v4f;

// fp32 -> bf16, round-to-nearest-even
static __device__ __forceinline__ __bf16 f2bf(float f) {
    uint32_t u = __builtin_bit_cast(uint32_t, f);
    uint32_t r = u + 0x7FFFu + ((u >> 16) & 1u);
    uint16_t h = (uint16_t)(r >> 16);
    return __builtin_bit_cast(__bf16, h);
}

// Hardware tanh on gfx1250 (TRANS32 op); branchless exp-based fallback.
static __device__ __forceinline__ float fast_tanh(float x) {
#if __has_builtin(__builtin_amdgcn_tanhf)
    return __builtin_amdgcn_tanhf(x);
#elif __has_builtin(__builtin_amdgcn_tanh_f32)
    return __builtin_amdgcn_tanh_f32(x);
#else
    // tanh(x) = 1 - 2/(exp(2x)+1); saturates correctly for |x| large
    float e = __expf(2.0f * x);
    float r = __builtin_amdgcn_rcpf(e + 1.0f);
    return fmaf(-2.0f, r, 1.0f);
#endif
}

// ---------------------------------------------------------------------------
// Kernel 0: pack W_enc (rows 512..1535 of W_transform, shape K=1024 x N=512,
// row-major over N) into bf16, transposed to column-major: Wt[n*1024 + k].
// Gives each WMMA lane two contiguous 16B loads per B-fragment.
// ---------------------------------------------------------------------------
__global__ void pack_wenc(const float* __restrict__ W, uint16_t* __restrict__ Wt_u16) {
    __bf16* Wt = (__bf16*)Wt_u16;
    int idx = blockIdx.x * blockDim.x + threadIdx.x;   // idx = n*ED + k
    int n = idx >> 10;
    int k = idx & (ED - 1);
    Wt[idx] = f2bf(W[(size_t)(DD + k) * DD + n]);
}

// ---------------------------------------------------------------------------
// Kernel 1: dec_proj[b,d] = decoder_state[b,:] @ W_dec[:,d] + bias[d]  (fp32)
// ---------------------------------------------------------------------------
__global__ void dec_proj_kernel(const float* __restrict__ dec, const float* __restrict__ W,
                                const float* __restrict__ bias, float* __restrict__ out) {
    int idx = blockIdx.x * blockDim.x + threadIdx.x;   // b*DD + d
    int b = idx >> 9;
    int d = idx & (DD - 1);
    float acc = bias[d];
    const float* drow = dec + (size_t)b * DD;
#pragma unroll 4
    for (int k = 0; k < DD; ++k)
        acc = fmaf(drow[k], W[(size_t)k * DD + d], acc);
    out[idx] = acc;
}

// ---------------------------------------------------------------------------
// Kernel 2: fused  tanh(enc @ W_enc + dec_proj) . v  -> masked scores (B,S)
// Block = 256 threads = 8 waves; covers 128 sequence rows of one batch.
// A slab (128x1024 fp32, contiguous) staged to LDS as bf16 (256 KB / WGP).
// Each wave: 16-row M-tile; per 16-col N-tile, 32 K-steps of
// v_wmma_f32_16x16x32_bf16 split across two accumulators for XDL ILP.
// ---------------------------------------------------------------------------
__global__ __launch_bounds__(256, 1)
void attn_scores(const float* __restrict__ enc, const uint16_t* __restrict__ Wt_u16,
                 const float* __restrict__ decp, const float* __restrict__ v,
                 const uint8_t* __restrict__ mask, float* __restrict__ scores) {
    extern __shared__ __bf16 smemA[];                  // 128 x 1024 bf16 = 256 KB
    const __bf16* Wt = (const __bf16*)Wt_u16;

    const int b   = blockIdx.y;
    const int s0  = blockIdx.x * 128;
    const int tid = threadIdx.x;

    // ---- stage A block: contiguous 512KB fp32 stream -> 256KB bf16 in LDS ----
    const v4f* src = (const v4f*)(enc + ((size_t)b * S_ + s0) * ED);
    for (int i = tid; i < (128 * ED) / 4; i += 256) {
        v4f f = src[i];
        v4bf p;
        p[0] = f2bf(f[0]); p[1] = f2bf(f[1]); p[2] = f2bf(f[2]); p[3] = f2bf(f[3]);
        *(v4bf*)(smemA + (size_t)i * 4) = p;
    }
    __syncthreads();

    const int wave = tid >> 5;
    const int lane = tid & 31;
    const int ln   = lane & 15;     // position within half-wave
    const int h    = lane >> 4;     // half select

    // A fragment row for this lane (16-bit A 16x32 layout, ISA 7.12.2)
    const __bf16* Arow = smemA + (size_t)(wave * 16 + ln) * ED;

    float sacc[8];
#pragma unroll
    for (int j = 0; j < 8; ++j) sacc[j] = 0.0f;

    for (int nt = 0; nt < DD / 16; ++nt) {
        const int n = nt * 16 + ln;                    // this lane's output column
        const __bf16* Bcol = Wt + (size_t)n * ED;
        v8f c0 = {}, c1 = {};                          // two chains -> XDL ILP
#pragma unroll 4
        for (int kc = 0; kc < ED / 32; kc += 2) {
            {
                const int k0 = kc * 32;
                v8bf a_lo = *(const v8bf*)(Arow + k0 + 8 * h);
                v8bf a_hi = *(const v8bf*)(Arow + k0 + 16 + 8 * h);
                v16bf a = __builtin_shufflevector(a_lo, a_hi,
                    0,1,2,3,4,5,6,7,8,9,10,11,12,13,14,15);
                v8bf b_lo = *(const v8bf*)(Bcol + k0 + 8 * h);
                v8bf b_hi = *(const v8bf*)(Bcol + k0 + 16 + 8 * h);
                v16bf bm = __builtin_shufflevector(b_lo, b_hi,
                    0,1,2,3,4,5,6,7,8,9,10,11,12,13,14,15);
                c0 = __builtin_amdgcn_wmma_f32_16x16x32_bf16(
                        false, a, false, bm, (short)0, c0, false, false);
            }
            {
                const int k0 = (kc + 1) * 32;
                v8bf a_lo = *(const v8bf*)(Arow + k0 + 8 * h);
                v8bf a_hi = *(const v8bf*)(Arow + k0 + 16 + 8 * h);
                v16bf a = __builtin_shufflevector(a_lo, a_hi,
                    0,1,2,3,4,5,6,7,8,9,10,11,12,13,14,15);
                v8bf b_lo = *(const v8bf*)(Bcol + k0 + 8 * h);
                v8bf b_hi = *(const v8bf*)(Bcol + k0 + 16 + 8 * h);
                v16bf bm = __builtin_shufflevector(b_lo, b_hi,
                    0,1,2,3,4,5,6,7,8,9,10,11,12,13,14,15);
                c1 = __builtin_amdgcn_wmma_f32_16x16x32_bf16(
                        false, a, false, bm, (short)0, c1, false, false);
            }
        }
        // fused epilogue: feats = tanh(c + dec_proj[n]); sacc += v[n]*feats
        const float dn = decp[b * DD + n];
        const float vn = v[n];
#pragma unroll
        for (int j = 0; j < 8; ++j)
            sacc[j] = fmaf(vn, fast_tanh(c0[j] + c1[j] + dn), sacc[j]);
    }

    // reduce over the 16 lanes of each half-wave (N dimension), then write
#pragma unroll
    for (int j = 0; j < 8; ++j) {
        float r = sacc[j];
        r += __shfl_xor(r, 1, 32);
        r += __shfl_xor(r, 2, 32);
        r += __shfl_xor(r, 4, 32);
        r += __shfl_xor(r, 8, 32);
        if (ln == 0) {
            const int sp  = s0 + wave * 16 + j + 8 * h;   // C layout: M=j (+8 for upper half)
            const int idx = b * S_ + sp;
            scores[idx] = mask[idx] ? r : NEG_INF;
        }
    }
}

// ---------------------------------------------------------------------------
// Kernel 3: row softmax over S=1024, one block per batch row
// ---------------------------------------------------------------------------
__global__ void softmax_kernel(const float* __restrict__ scores, float* __restrict__ out) {
    __shared__ float red[256];
    const int b = blockIdx.x;
    const int tid = threadIdx.x;
    float lv[4];
    float mx = -3.4e38f;
#pragma unroll
    for (int i = 0; i < 4; ++i) {
        lv[i] = scores[b * S_ + tid + i * 256];
        mx = fmaxf(mx, lv[i]);
    }
    red[tid] = mx; __syncthreads();
    for (int off = 128; off > 0; off >>= 1) {
        if (tid < off) red[tid] = fmaxf(red[tid], red[tid + off]);
        __syncthreads();
    }
    mx = red[0]; __syncthreads();
    float sum = 0.0f;
#pragma unroll
    for (int i = 0; i < 4; ++i) {
        lv[i] = __expf(lv[i] - mx);
        sum += lv[i];
    }
    red[tid] = sum; __syncthreads();
    for (int off = 128; off > 0; off >>= 1) {
        if (tid < off) red[tid] += red[tid + off];
        __syncthreads();
    }
    const float inv = 1.0f / red[0];
#pragma unroll
    for (int i = 0; i < 4; ++i)
        out[b * S_ + tid + i * 256] = lv[i] * inv;
}

// ---------------------------------------------------------------------------
extern "C" void kernel_launch(void* const* d_in, const int* in_sizes, int n_in,
                              void* d_out, int out_size, void* d_ws, size_t ws_size,
                              hipStream_t stream) {
    const float*   dec  = (const float*)d_in[0];    // (B, DD)
    const float*   enc  = (const float*)d_in[1];    // (B, S, ED)
    const uint8_t* mask = (const uint8_t*)d_in[2];  // (B, S) bool (1 byte)
    const float*   W    = (const float*)d_in[3];    // (ED+DD, DD)
    const float*   bias = (const float*)d_in[4];    // (DD,)
    const float*   v    = (const float*)d_in[5];    // (DD,)
    float* out = (float*)d_out;

    char* ws = (char*)d_ws;
    float*    decp   = (float*)ws;                           // 64*512 f32   = 128 KB
    float*    scores = (float*)(ws + 131072);                // 64*1024 f32  = 256 KB
    uint16_t* Wt     = (uint16_t*)(ws + 131072 + 262144);    // 512*1024 bf16 = 1 MB

    pack_wenc<<<(DD * ED) / 256, 256, 0, stream>>>(W, Wt);
    dec_proj_kernel<<<(B_ * DD) / 256, 256, 0, stream>>>(dec, W, bias, decp);
    attn_scores<<<dim3(S_ / 128, B_), 256, 128 * ED * sizeof(uint16_t), stream>>>(
        enc, Wt, decp, v, mask, scores);
    softmax_kernel<<<B_, 256, 0, stream>>>(scores, out);
}
